// LinearAttention_61452392071236
// MI455X (gfx1250) — matrix operand order repaired and checked by reading, not compiled, not run on hardware
//
#include <hip/hip_runtime.h>
#include <math.h>

// ---- problem constants (from reference) ----
#define B_    8
#define CIN   64
#define NH    4
#define DH    32
#define NPIX  81920      /* 80*1024 */
#define OC3   384        /* 3*NH*DH */
#define CE    128        /* NH*DH   */
#define COUT  64
#define TN    64         /* pixel tile */

typedef __attribute__((ext_vector_type(16))) __bf16 v16bf;
typedef __attribute__((ext_vector_type(8)))  float  v8f;
typedef __attribute__((ext_vector_type(4)))  int    v4i_;

union BF16Frag { v16bf v; unsigned short s[16]; unsigned int u[8]; };

// native f32->bf16 convert (v_cvt_*bf16 on gfx1250, RNE)
__device__ __forceinline__ unsigned short f2bf(float f) {
  __bf16 h = (__bf16)f;
  return __builtin_bit_cast(unsigned short, h);
}
__device__ __forceinline__ float bf2f(unsigned short h) {
  return __uint_as_float(((unsigned int)h) << 16);
}
// A-fragment K offset for 16-bit 16x16x32 WMMA (ISA 7.12.2 table)
__device__ __forceinline__ int kbaseA(int j, int half) {
  return ((j >> 2) << 4) + (half << 3) + ((j & 3) << 1);
}

// ---- gfx1250 async global->LDS copy (ASYNCcnt path), with safe fallback ----
#if defined(__has_builtin)
#if __has_builtin(__builtin_amdgcn_global_load_async_to_lds_b128)
#define HAVE_ASYNC_LDS 1
#endif
#endif
#ifndef HAVE_ASYNC_LDS
#define HAVE_ASYNC_LDS 0
#endif

typedef __attribute__((address_space(1))) v4i_* gv4p;   // global int4*
typedef __attribute__((address_space(3))) v4i_* lv4p;   // LDS    int4*

__device__ __forceinline__ void async_copy16(const void* g, void* l) {
#if HAVE_ASYNC_LDS
  __builtin_amdgcn_global_load_async_to_lds_b128((gv4p)(g), (lv4p)(l), 0, 0);
#else
  *(uint4*)l = *(const uint4*)g;
#endif
}
__device__ __forceinline__ void wait_async0() {
#if defined(__has_builtin)
#if __has_builtin(__builtin_amdgcn_s_wait_asynccnt)
  __builtin_amdgcn_s_wait_asynccnt(0);
  return;
#endif
#endif
  asm volatile("s_wait_asynccnt 0x0" ::: "memory");
}

// ============================================================
// Kernel 0: one-time weight conversion to bf16 (L2-resident)
// ============================================================
__global__ __launch_bounds__(256) void prep_weights_kernel(
    const float* __restrict__ w_qkv, const float* __restrict__ w_out,
    unsigned short* __restrict__ wbf, unsigned short* __restrict__ wobf) {
  int tid = blockIdx.x * 256 + threadIdx.x;
  if (tid < OC3 * CIN) wbf[tid] = f2bf(w_qkv[tid]);
  if (tid < COUT * CE) wobf[tid] = f2bf(w_out[tid]);
}

// ============================================================
// Kernel 1: qkv[b,o,p] = sum_c w_qkv[o,c] * x[b,c,p]
// M=384 (24 tiles), K=64 (2 wmma steps), N tile = 64 pixels.
// X tile async-copied to LDS; all 4 B subtiles held in registers;
// A loaded once per M tile; results staged in LDS, written b128.
// ============================================================
__global__ __launch_bounds__(256) void qkv_proj_kernel(
    const float* __restrict__ x, const unsigned short* __restrict__ wbf,
    unsigned short* __restrict__ qw, unsigned short* __restrict__ kw,
    unsigned short* __restrict__ vw) {
  __shared__ __attribute__((aligned(16))) float          Xf[CIN * TN];    // 16KB (raw fp32)
  __shared__ __attribute__((aligned(16))) unsigned short Olds[OC3 * TN];  // 48KB (bf16 out stage)
  const int tid = threadIdx.x;
  const int b   = blockIdx.y;
  const int p0  = blockIdx.x * TN;

  // async copy of the X tile (raw fp32, contiguous 16B on both sides)
  const float* xb = x + ((size_t)b * CIN) * NPIX + p0;
  for (int i = tid * 4; i < CIN * TN; i += 256 * 4) {
    int c = i >> 6, p = i & 63;
    async_copy16(xb + (size_t)c * NPIX + p, &Xf[i]);
  }
  wait_async0();
  __syncthreads();

  const int wave = tid >> 5, lane = tid & 31;
  const int half = lane >> 4, l16 = lane & 15;

  // preload all four B subtiles (K=64 -> 2 fragments each)
  BF16Frag bfr[4][2];
  #pragma unroll
  for (int ns = 0; ns < 4; ++ns) {
    const int pix = ns * 16 + l16;
    #pragma unroll
    for (int g = 0; g < 2; ++g)
      #pragma unroll
      for (int j = 0; j < 8; ++j) {
        int kk = g * 32 + kbaseA(j, half);
        bfr[ns][g].s[2 * j]     = f2bf(Xf[kk * TN + pix]);
        bfr[ns][g].s[2 * j + 1] = f2bf(Xf[(kk + 1) * TN + pix]);
      }
  }

  for (int mt = wave; mt < 24; mt += 8) {
    BF16Frag af[2];
    const unsigned short* wr = wbf + (mt * 16 + l16) * CIN;
    #pragma unroll
    for (int g = 0; g < 2; ++g)
      #pragma unroll
      for (int j = 0; j < 8; ++j)
        af[g].u[j] = *(const unsigned int*)&wr[g * 32 + kbaseA(j, half)];
    #pragma unroll
    for (int ns = 0; ns < 4; ++ns) {
      v8f acc = {};
      acc = __builtin_amdgcn_wmma_f32_16x16x32_bf16(false, af[0].v, false, bfr[ns][0].v,
                                                    (short)0, acc, false, false);
      acc = __builtin_amdgcn_wmma_f32_16x16x32_bf16(false, af[1].v, false, bfr[ns][1].v,
                                                    (short)0, acc, false, false);
      const int pix = ns * 16 + l16;
      #pragma unroll
      for (int r = 0; r < 8; ++r)
        Olds[(mt * 16 + half * 8 + r) * TN + pix] = f2bf(acc[r]);
    }
  }
  __syncthreads();

  // coalesced write-out: 16B (8 bf16 pixels) per lane, uniform q/k/v select
  const uint4* o4 = (const uint4*)Olds;
  for (int i = tid; i < (OC3 * TN) / 8; i += 256) {
    int row = i >> 3, seg = i & 7;
    unsigned short* dst; int ch;
    if (row < CE)          { dst = qw; ch = row; }
    else if (row < 2 * CE) { dst = kw; ch = row - CE; }
    else                   { dst = vw; ch = row - 2 * CE; }
    *(uint4*)&dst[((size_t)b * CE + ch) * NPIX + p0 + seg * 8] = o4[i];
  }
}

// ============================================================
// Kernel 2: per-row (b*128+ch) max and 1/sum(exp) over n=81920
// Packed-uint streaming reads (2 bf16 per load).
// ============================================================
__global__ __launch_bounds__(256) void softmax_stats_kernel(
    const unsigned short* __restrict__ kw, float* __restrict__ stats) {
  __shared__ float red[256];
  const int row = blockIdx.x;
  const unsigned int* rp = (const unsigned int*)(kw + (size_t)row * NPIX);
  const int tid = threadIdx.x;

  float m = -3.4e38f;
  for (int i = tid; i < NPIX / 2; i += 256) {
    __builtin_prefetch(rp + i + 2048, 0, 0);   // global_prefetch_b8, streaming
    unsigned int u = rp[i];
    m = fmaxf(m, fmaxf(bf2f((unsigned short)u), bf2f((unsigned short)(u >> 16))));
  }
  red[tid] = m; __syncthreads();
  for (int s = 128; s > 0; s >>= 1) {
    if (tid < s) red[tid] = fmaxf(red[tid], red[tid + s]);
    __syncthreads();
  }
  m = red[0]; __syncthreads();

  float sum = 0.f;
  for (int i = tid; i < NPIX / 2; i += 256) {
    unsigned int u = rp[i];
    sum += __expf(bf2f((unsigned short)u) - m) +
           __expf(bf2f((unsigned short)(u >> 16)) - m);
  }
  red[tid] = sum; __syncthreads();
  for (int s = 128; s > 0; s >>= 1) {
    if (tid < s) red[tid] += red[tid + s];
    __syncthreads();
  }
  if (tid == 0) { stats[row * 2] = m; stats[row * 2 + 1] = 1.0f / red[0]; }
}

// ============================================================
// Kernel 3: context[b,h,d,e] = (1/sum_d) * sum_n exp(k[d,n]-max_d) * v[e,n]
// One block per (b,h); 8 waves split the 81920-long contraction,
// each accumulates 2x2 wmma tiles, LDS-reduce at the end.
// ============================================================
__global__ __launch_bounds__(256) void context_kernel(
    const unsigned short* __restrict__ kw, const unsigned short* __restrict__ vw,
    const float* __restrict__ stats, float* __restrict__ ctx) {
  __shared__ float cred[8 * DH * DH];  // 32KB
  const int bh = blockIdx.x;           // b*4 + h
  const int b = bh >> 2, h = bh & 3;
  const int tid = threadIdx.x, wave = tid >> 5, lane = tid & 31;
  const int half = lane >> 4, l16 = lane & 15;
  const size_t base = ((size_t)b * CE + h * DH) * NPIX;

  float mx[2];
  mx[0] = stats[(bh * DH + l16) * 2];
  mx[1] = stats[(bh * DH + 16 + l16) * 2];

  v8f acc[2][2] = {};
  for (int nc = wave * 32; nc < NPIX; nc += 256) {
    BF16Frag af[2], bf[2];
    #pragma unroll
    for (int mt = 0; mt < 2; ++mt) {
      const unsigned short* kr = kw + base + (size_t)(mt * 16 + l16) * NPIX + nc;
      #pragma unroll
      for (int j = 0; j < 8; ++j) {
        unsigned int pr = *(const unsigned int*)&kr[kbaseA(j, half)];
        af[mt].s[2 * j]     = f2bf(__expf(bf2f((unsigned short)(pr & 0xFFFFu)) - mx[mt]));
        af[mt].s[2 * j + 1] = f2bf(__expf(bf2f((unsigned short)(pr >> 16)) - mx[mt]));
      }
    }
    #pragma unroll
    for (int nt = 0; nt < 2; ++nt) {
      const unsigned short* vr = vw + base + (size_t)(nt * 16 + l16) * NPIX + nc;
      #pragma unroll
      for (int j = 0; j < 8; ++j)
        bf[nt].u[j] = *(const unsigned int*)&vr[kbaseA(j, half)];
    }
    #pragma unroll
    for (int mt = 0; mt < 2; ++mt)
      #pragma unroll
      for (int nt = 0; nt < 2; ++nt)
        acc[mt][nt] = __builtin_amdgcn_wmma_f32_16x16x32_bf16(
            false, af[mt].v, false, bf[nt].v, (short)0, acc[mt][nt], false, false);
  }

  #pragma unroll
  for (int mt = 0; mt < 2; ++mt)
    #pragma unroll
    for (int nt = 0; nt < 2; ++nt)
      #pragma unroll
      for (int r = 0; r < 8; ++r) {
        int d = mt * 16 + half * 8 + r, e = nt * 16 + l16;
        cred[(wave * DH + d) * DH + e] = acc[mt][nt][r];
      }
  __syncthreads();
  for (int i = tid; i < DH * DH; i += 256) {
    int d = i >> 5;
    float s = 0.f;
    #pragma unroll
    for (int w2 = 0; w2 < 8; ++w2) s += cred[w2 * DH * DH + i];
    ctx[(size_t)bh * (DH * DH) + i] = s * stats[(bh * DH + d) * 2 + 1];
  }
}

// ============================================================
// Kernel 4 (fused): out[eG,p] = sum_d ctx[d,e]*q[d,p]  (staged in LDS, bf16)
//            then   final[o,p] = sum_e w_out[o,e]*out[e,p] + b_out[o]
// ============================================================
__global__ __launch_bounds__(256) void out_proj_kernel(
    const unsigned short* __restrict__ qw, const float* __restrict__ ctx,
    const unsigned short* __restrict__ wobf, const float* __restrict__ b_out,
    float* __restrict__ out) {
  __shared__ __attribute__((aligned(16))) unsigned short ctxlds[NH * DH * DH]; //  8KB  [h][e*32+d] (transposed)
  __shared__ __attribute__((aligned(16))) unsigned short qlds[CE * TN];        // 16KB
  __shared__ __attribute__((aligned(16))) unsigned short wolds[COUT * CE];     // 16KB
  __shared__ __attribute__((aligned(16))) unsigned short outlds[CE * TN];      // 16KB
  __shared__ float bias[COUT];
  const int tid = threadIdx.x;
  const int b   = blockIdx.y;
  const int p0  = blockIdx.x * TN;

  // async raw copies: q tile and bf16 w_out
  const unsigned short* qb = qw + ((size_t)b * CE) * NPIX + p0;
  for (int i = tid * 8; i < CE * TN; i += 256 * 8) {
    int c = i >> 6, p = i & 63;
    async_copy16(qb + (size_t)c * NPIX + p, &qlds[i]);
  }
  for (int i = tid * 8; i < COUT * CE; i += 256 * 8)
    async_copy16(&wobf[i], &wolds[i]);

  // converted / transposed fills
  for (int i = tid; i < NH * DH * DH; i += 256) {
    int hh = i >> 10, r2 = i & 1023, e = r2 >> 5, d = r2 & 31;
    ctxlds[i] = f2bf(ctx[(size_t)(b * NH + hh) * (DH * DH) + d * DH + e]);
  }
  if (tid < COUT) bias[tid] = b_out[tid];
  wait_async0();
  __syncthreads();

  const int wave = tid >> 5, lane = tid & 31;
  const int half = lane >> 4, l16 = lane & 15;

  // Phase A: 8 M-tiles (eG) x 4 N-subtiles, K=32 per head -> 1 wmma each
  for (int t = wave; t < 32; t += 8) {
    int em = t >> 2, ns = t & 3, hh = em >> 1;
    int e_loc = (em & 1) * 16 + l16;
    int pix = ns * 16 + l16;
    BF16Frag af, bf;
    #pragma unroll
    for (int j = 0; j < 8; ++j)
      af.u[j] = *(const unsigned int*)&ctxlds[(hh * DH + e_loc) * DH + kbaseA(j, half)];
    #pragma unroll
    for (int j = 0; j < 8; ++j) {
      int kk = kbaseA(j, half);
      bf.s[2 * j]     = qlds[(hh * DH + kk) * TN + pix];
      bf.s[2 * j + 1] = qlds[(hh * DH + kk + 1) * TN + pix];
    }
    v8f acc = {};
    acc = __builtin_amdgcn_wmma_f32_16x16x32_bf16(false, af.v, false, bf.v,
                                                  (short)0, acc, false, false);
    #pragma unroll
    for (int r = 0; r < 8; ++r)
      outlds[(em * 16 + half * 8 + r) * TN + pix] = f2bf(acc[r]);
  }
  __syncthreads();

  // Phase B: 4 M-tiles (o) x 4 N-subtiles, K=128 -> 4 wmma steps
  for (int t = wave; t < 16; t += 8) {
    int ot = t >> 2, ns = t & 3;
    int pix = ns * 16 + l16;
    v8f acc = {};
    #pragma unroll
    for (int g = 0; g < 4; ++g) {
      BF16Frag af, bf;
      #pragma unroll
      for (int j = 0; j < 8; ++j)
        af.u[j] = *(const unsigned int*)&wolds[(ot * 16 + l16) * CE + g * 32 + kbaseA(j, half)];
      #pragma unroll
      for (int j = 0; j < 8; ++j) {
        int kk = g * 32 + kbaseA(j, half);
        bf.s[2 * j]     = outlds[kk * TN + pix];
        bf.s[2 * j + 1] = outlds[(kk + 1) * TN + pix];
      }
      acc = __builtin_amdgcn_wmma_f32_16x16x32_bf16(false, af.v, false, bf.v,
                                                    (short)0, acc, false, false);
    }
    #pragma unroll
    for (int r = 0; r < 8; ++r) {
      int o = ot * 16 + half * 8 + r;
      out[((size_t)b * COUT + o) * NPIX + p0 + pix] = acc[r] + bias[o];
    }
  }
}

// ============================================================
extern "C" void kernel_launch(void* const* d_in, const int* in_sizes, int n_in,
                              void* d_out, int out_size, void* d_ws, size_t ws_size,
                              hipStream_t stream) {
  const float* x     = (const float*)d_in[0];
  const float* w_qkv = (const float*)d_in[1];
  const float* w_out = (const float*)d_in[2];
  const float* b_out = (const float*)d_in[3];
  float* out = (float*)d_out;

  // workspace layout: bf16 q/k/v (3x160MB), softmax stats, context, bf16 weights
  const size_t qkvElems = (size_t)B_ * CE * NPIX;
  unsigned short* qw = (unsigned short*)d_ws;
  unsigned short* kw = qw + qkvElems;
  unsigned short* vw = kw + qkvElems;
  float* stats = (float*)(vw + qkvElems);                 // [1024][2]: max, 1/sum
  float* ctxp  = stats + (size_t)B_ * CE * 2;             // [32][32*32]
  unsigned short* wbf  = (unsigned short*)(ctxp + (size_t)B_ * NH * DH * DH);
  unsigned short* wobf = wbf + OC3 * CIN;

  dim3 blk(256);
  dim3 gTile(NPIX / TN, B_);
  hipLaunchKernelGGL(prep_weights_kernel, dim3(96), blk, 0, stream, w_qkv, w_out, wbf, wobf);
  hipLaunchKernelGGL(qkv_proj_kernel, gTile, blk, 0, stream, x, wbf, qw, kw, vw);
  hipLaunchKernelGGL(softmax_stats_kernel, dim3(B_ * CE), blk, 0, stream, kw, stats);
  hipLaunchKernelGGL(context_kernel, dim3(B_ * NH), blk, 0, stream, kw, vw, stats, ctxp);
  hipLaunchKernelGGL(out_proj_kernel, gTile, blk, 0, stream, qw, ctxp, wobf, b_out, out);
}
